// lr_cls1_84610855731251
// MI455X (gfx1250) — matrix operand order repaired and checked by reading, not compiled, not run on hardware
//
#include <hip/hip_runtime.h>
#include <stdint.h>

// ---------------------------------------------------------------------------
// 3-layer MLP on MI455X (gfx1250):
//   Pass 0: convert X, W1, W2, W3 fp32 -> bf16 once (into d_ws).
//   Passes 1-3: bf16 WMMA GEMMs  C = act(A * W^T + b), fp32 accumulate.
// GEMM tile 128x128x32, 256 threads (8 wave32), wave tile 64x32 (4x2 WMMA).
// Global->LDS staging uses GLOBAL_LOAD_ASYNC_TO_LDS_B128 (ASYNCcnt),
// double-buffered so the async copy of tile k+1 overlaps WMMA on tile k.
// ---------------------------------------------------------------------------

typedef __attribute__((ext_vector_type(16))) __bf16 v16bf;
typedef __attribute__((ext_vector_type(8)))  float  v8f;

#define BM 128
#define BN 128
#define BK 32
#define LDK (BK + 8)                 // 40 elems = 80B row stride: conflict-free perm
#define TILE_USHORTS (BM * LDK)      // 5120 elems per tile buffer

__device__ __forceinline__ unsigned short f2bf(float f) {
    unsigned int u = __float_as_uint(f);
    u += 0x7FFFu + ((u >> 16) & 1u);          // round-to-nearest-even
    return (unsigned short)(u >> 16);
}
__device__ __forceinline__ unsigned pk2(float x, float y) {
    return (unsigned)f2bf(x) | ((unsigned)f2bf(y) << 16);
}

union Frag16 { v16bf v; uint4 q[2]; };

// ---- pass 0: elementwise fp32 -> bf16 (8 elements / thread) ----------------
__global__ __launch_bounds__(256)
void cvt_f32_bf16(const float* __restrict__ s, unsigned short* __restrict__ d, int n8) {
    int i = blockIdx.x * 256 + threadIdx.x;
    if (i >= n8) return;
    const float4* sp = (const float4*)s + (size_t)i * 2;
    float4 a = sp[0], b = sp[1];
    uint4 p;
    p.x = pk2(a.x, a.y); p.y = pk2(a.z, a.w);
    p.z = pk2(b.x, b.y); p.w = pk2(b.z, b.w);
    ((uint4*)d)[i] = p;
}

// ---- bf16 GEMM with async global->LDS double buffering ---------------------
template<bool RELU, bool OUT_BF16>
__global__ __launch_bounds__(256)
void mlp_gemm(const unsigned short* __restrict__ A, const unsigned short* __restrict__ Wb,
              const float* __restrict__ bias, void* __restrict__ Cptr,
              int M, int N, int K)
{
    __shared__ unsigned short ldsA[2 * TILE_USHORTS];   // ping-pong A tiles
    __shared__ unsigned short ldsB[2 * TILE_USHORTS];   // ping-pong W tiles

    const int tid  = threadIdx.x;
    const int lane = tid & 31;
    const int wave = tid >> 5;       // 0..7
    const int wM   = wave >> 2;      // 0..1 -> 64 rows
    const int wN   = wave & 3;       // 0..3 -> 32 cols
    const int half = lane >> 4;
    const int l16  = lane & 15;

    const int m0 = blockIdx.y * BM;
    const int n0 = blockIdx.x * BN;

    // wave-relative LDS byte offsets (addr[31:0] of the generic pointer)
    const unsigned ldsAoff = (unsigned)(uintptr_t)&ldsA[0];
    const unsigned ldsBoff = (unsigned)(uintptr_t)&ldsB[0];

    // Stage one 128x32 bf16 tile of A and W into LDS buffer `buf` via the
    // CDNA5 async global->LDS path. 512 16B chunks per tile, 2 per thread.
    auto stage = [&](int k0, int buf) {
#pragma unroll
        for (int i = 0; i < 2; ++i) {
            int linear = tid + i * 256;          // 0..511
            int row = linear >> 2;               // 0..127
            int kc  = linear & 3;                // 4 chunks of 8 bf16 per row
            unsigned ldst = (unsigned)(row * LDK + kc * 8) * 2u
                          + (unsigned)buf * (TILE_USHORTS * 2u);
            unsigned loA = ldsAoff + ldst;
            unsigned loB = ldsBoff + ldst;
            uint64_t gA = (uint64_t)(uintptr_t)(A  + (size_t)(m0 + row) * K + k0 + kc * 8);
            uint64_t gB = (uint64_t)(uintptr_t)(Wb + (size_t)(n0 + row) * K + k0 + kc * 8);
            asm volatile("global_load_async_to_lds_b128 %0, %1, off"
                         :: "v"(loA), "v"(gA) : "memory");
            asm volatile("global_load_async_to_lds_b128 %0, %1, off"
                         :: "v"(loB), "v"(gB) : "memory");
        }
    };

    v8f acc[4][2];
#pragma unroll
    for (int i = 0; i < 4; ++i)
#pragma unroll
        for (int j = 0; j < 2; ++j)
#pragma unroll
            for (int e = 0; e < 8; ++e) acc[i][j][e] = 0.0f;

    stage(0, 0);
    asm volatile("s_wait_asynccnt 0x0" ::: "memory");
    __syncthreads();

    const int nsteps = K / BK;
    for (int s = 0; s < nsteps; ++s) {
        const int buf = s & 1;
        if (s + 1 < nsteps) stage((s + 1) * BK, buf ^ 1);   // prefetch next tile

        const unsigned short* tA = &ldsA[buf * TILE_USHORTS];
        const unsigned short* tB = &ldsB[buf * TILE_USHORTS];

        Frag16 bfr[2];
#pragma unroll
        for (int nt = 0; nt < 2; ++nt) {
            // B(=W^T) col tile: lane holds W row n, lanes 0-15 K={0..7,16..23}, lanes 16-31 K={8..15,24..31}
            int nrow = wN * 32 + nt * 16 + l16;
            const unsigned short* src = &tB[nrow * LDK + half * 8];
            bfr[nt].q[0] = *(const uint4*)(src);
            bfr[nt].q[1] = *(const uint4*)(src + 16);
        }
#pragma unroll
        for (int mt = 0; mt < 4; ++mt) {
            // A frag: lanes 0-15 K={0..7,16..23}, lanes 16-31 K={8..15,24..31}
            int mrow = wM * 64 + mt * 16 + l16;
            const unsigned short* sA = &tA[mrow * LDK + half * 8];
            Frag16 a;
            a.q[0] = *(const uint4*)(sA);
            a.q[1] = *(const uint4*)(sA + 16);
#pragma unroll
            for (int nt = 0; nt < 2; ++nt) {
                acc[mt][nt] = __builtin_amdgcn_wmma_f32_16x16x32_bf16(
                    false, a.v, false, bfr[nt].v,
                    (short)0, acc[mt][nt], false, false);
            }
        }
        // async prefetch issued before compute must land before next iteration
        asm volatile("s_wait_asynccnt 0x0" ::: "memory");
        __syncthreads();
    }

    // ---- epilogue: bias (+ReLU), store bf16 hidden or fp32 final ----
#pragma unroll
    for (int nt = 0; nt < 2; ++nt) {
        int n = n0 + wN * 32 + nt * 16 + l16;       // C/D layout: N = lane&15
        float bv = bias[n];
#pragma unroll
        for (int mt = 0; mt < 4; ++mt) {
#pragma unroll
            for (int r = 0; r < 8; ++r) {
                int m = m0 + wM * 64 + mt * 16 + half * 8 + r;  // VGPR r -> M=r(+8)
                float v = acc[mt][nt][r] + bv;
                if (RELU) v = v > 0.0f ? v : 0.0f;
                if (OUT_BF16)
                    ((unsigned short*)Cptr)[(size_t)m * N + n] = f2bf(v);
                else
                    ((float*)Cptr)[(size_t)m * N + n] = v;
            }
        }
    }
}

extern "C" void kernel_launch(void* const* d_in, const int* in_sizes, int n_in,
                              void* d_out, int out_size, void* d_ws, size_t ws_size,
                              hipStream_t stream) {
    (void)in_sizes; (void)n_in; (void)out_size; (void)ws_size;
    const float* X  = (const float*)d_in[0];
    const float* W1 = (const float*)d_in[1];
    const float* b1 = (const float*)d_in[2];
    const float* W2 = (const float*)d_in[3];
    const float* b2 = (const float*)d_in[4];
    const float* W3 = (const float*)d_in[5];
    const float* b3 = (const float*)d_in[6];
    float* out = (float*)d_out;

    const int M = 32 * 256;                       // 8192
    const size_t nX  = (size_t)M * 2048;
    const size_t nW1 = (size_t)4096 * 2048;
    const size_t nW2 = (size_t)4096 * 4096;
    const size_t nW3 = (size_t)2048 * 4096;
    const size_t nH  = (size_t)M * 4096;

    unsigned short* ws  = (unsigned short*)d_ws;
    unsigned short* Xb  = ws;            ws += nX;
    unsigned short* W1b = ws;            ws += nW1;
    unsigned short* W2b = ws;            ws += nW2;
    unsigned short* W3b = ws;            ws += nW3;
    unsigned short* H1  = ws;            ws += nH;
    unsigned short* H2  = ws;

    dim3 blk(256);
    // Pass 0: one-time fp32 -> bf16 conversion of inputs + weights
    cvt_f32_bf16<<<dim3((unsigned)(nX  / 8 / 256)), blk, 0, stream>>>(X,  Xb,  (int)(nX  / 8));
    cvt_f32_bf16<<<dim3((unsigned)(nW1 / 8 / 256)), blk, 0, stream>>>(W1, W1b, (int)(nW1 / 8));
    cvt_f32_bf16<<<dim3((unsigned)(nW2 / 8 / 256)), blk, 0, stream>>>(W2, W2b, (int)(nW2 / 8));
    cvt_f32_bf16<<<dim3((unsigned)(nW3 / 8 / 256)), blk, 0, stream>>>(W3, W3b, (int)(nW3 / 8));

    // Layer 1: bf16 in, bf16 hidden out, ReLU
    mlp_gemm<true,  true ><<<dim3(4096 / BN, M / BM), blk, 0, stream>>>(
        Xb, W1b, b1, H1, M, 4096, 2048);
    // Layer 2
    mlp_gemm<true,  true ><<<dim3(4096 / BN, M / BM), blk, 0, stream>>>(
        H1, W2b, b2, H2, M, 4096, 4096);
    // Layer 3: fp32 out, no ReLU
    mlp_gemm<false, false><<<dim3(2048 / BN, M / BM), blk, 0, stream>>>(
        H2, W3b, b3, out, M, 2048, 4096);
}
